// BondGNN_78013785964684
// MI455X (gfx1250) — matrix operand order repaired and checked by reading, not compiled, not run on hardware
//
#include <hip/hip_runtime.h>

typedef __attribute__((ext_vector_type(16))) _Float16 v16h;
typedef __attribute__((ext_vector_type(8)))  float    v8f;

#define HID 64

// ---------------- utility kernels ----------------

__global__ void fill_kernel(float* __restrict__ p, float v, long long n) {
  long long i = (long long)blockIdx.x * blockDim.x + threadIdx.x;
  if (i < n) p[i] = v;
}

__global__ void deg_scatter_kernel(const int* __restrict__ dst,
                                   float* __restrict__ deg, long long E) {
  long long e = (long long)blockIdx.x * blockDim.x + threadIdx.x;
  if (e < E) atomicAdd(&deg[dst[e]], 1.0f);
}

__global__ void rsqrt_kernel(float* __restrict__ deg, int n) {
  int i = blockIdx.x * blockDim.x + threadIdx.x;
  if (i < n) deg[i] = rsqrtf(deg[i]);   // deg >= 1 always (self-loop)
}

// Transpose + convert weights once: W[K x 64] f32 -> Wt[64 x K] f16.
// Makes every WMMA B-fragment (one column, 16 consecutive K) a contiguous 32B load.
__global__ void convert_wt_kernel(const float* __restrict__ W,
                                  _Float16* __restrict__ Wt, int K) {
  int i = blockIdx.x * blockDim.x + threadIdx.x;
  if (i >= HID * K) return;
  int n = i / K, k = i % K;
  Wt[(size_t)n * K + k] = (_Float16)W[(size_t)k * HID + n];
}

// ---------------- WMMA GEMM: C[nrows x 64] = A[nrows x K] @ W[K x 64] ----------------
// One wave per 16-row strip; 4 accumulator tiles cover the 64 output columns.
// A: f32 loaded as b128 pairs, packed to f16 via v_cvt_pk_f16_f32.
// B: pre-transposed f16 (Wt), fragment = 32 contiguous bytes -> 2x global_load_b128.

__global__ void gemm_wmma_kernel(const float* __restrict__ A,
                                 const _Float16* __restrict__ Wt,
                                 float* __restrict__ C, int nrows, int K) {
  int gwave = (int)((blockIdx.x * (long long)blockDim.x + threadIdx.x) >> 5);
  int lane  = threadIdx.x & 31;
  int row0  = gwave * 16;
  if (row0 >= nrows) return;              // wave-uniform: EXEC stays all-1s for WMMA

  int g  = lane >> 4;                     // lane group (0: lanes 0-15, 1: lanes 16-31)
  int lm = lane & 15;                     // M (for A) / N (for B, C/D) within group

  v8f acc0 = {}, acc1 = {}, acc2 = {}, acc3 = {};

  int arow = row0 + lm;
  if (arow >= nrows) arow = nrows - 1;    // safety clamp (stores are guarded)
  const float* Ar = A + (size_t)arow * K;

  for (int kb = 0; kb < K; kb += 32) {
    // A fragment: lane holds K = kb+g*16 .. +15 (matches B fragment order)
    const float4* Ap0 = (const float4*)(Ar + kb + g * 16);
    const float4* Ap1 = (const float4*)(Ar + kb + g * 16 + 8);
    float4 a0 = Ap0[0], a1 = Ap0[1];
    float4 a2 = Ap1[0], a3 = Ap1[1];
    v16h a;
    a[0]  = (_Float16)a0.x; a[1]  = (_Float16)a0.y;
    a[2]  = (_Float16)a0.z; a[3]  = (_Float16)a0.w;
    a[4]  = (_Float16)a1.x; a[5]  = (_Float16)a1.y;
    a[6]  = (_Float16)a1.z; a[7]  = (_Float16)a1.w;
    a[8]  = (_Float16)a2.x; a[9]  = (_Float16)a2.y;
    a[10] = (_Float16)a2.z; a[11] = (_Float16)a2.w;
    a[12] = (_Float16)a3.x; a[13] = (_Float16)a3.y;
    a[14] = (_Float16)a3.z; a[15] = (_Float16)a3.w;

#pragma unroll
    for (int nt = 0; nt < 4; ++nt) {
      // B fragment: column nt*16+lm, K = kb+g*16 .. +15 -> 16 contiguous f16
      v16h b = *(const v16h*)(Wt + (size_t)(nt * 16 + lm) * K + kb + g * 16);
      v8f* accp = (nt == 0) ? &acc0 : (nt == 1) ? &acc1 : (nt == 2) ? &acc2 : &acc3;
      *accp = __builtin_amdgcn_wmma_f32_16x16x32_f16(
          /*neg_a=*/false, a, /*neg_b=*/false, b,
          /*c_mod=*/(short)0, *accp, /*reuse_a=*/false, /*reuse_b=*/false);
    }
  }

  // C/D layout: VGPR r -> row = row0 + g*8 + r, col = nt*16 + lm
#pragma unroll
  for (int nt = 0; nt < 4; ++nt) {
    v8f acc = (nt == 0) ? acc0 : (nt == 1) ? acc1 : (nt == 2) ? acc2 : acc3;
#pragma unroll
    for (int r = 0; r < 8; ++r) {
      int row = row0 + g * 8 + r;
      if (row < nrows) C[(size_t)row * HID + nt * 16 + lm] = acc[r];
    }
  }
}

// ---------------- edge aggregation: agg[dst] += h[src] * isq[src]*isq[dst] ----------------
// One thread per (edge, 16-feature chunk): b128 gathers, 16 f32 atomics (L2-resident).

__global__ void agg_kernel(const float* __restrict__ h,
                           const int* __restrict__ src, const int* __restrict__ dst,
                           const float* __restrict__ isq,
                           float* __restrict__ agg, long long nE, int nN) {
  long long idx = (long long)blockIdx.x * blockDim.x + threadIdx.x;
  long long e = idx >> 2;
  int c = (int)(idx & 3);
  if (e >= nE + (long long)nN) return;
  int s, d;
  if (e < nE) { s = src[e]; d = dst[e]; }
  else        { s = d = (int)(e - nE); }          // self-loop
  float norm = isq[s] * isq[d];
  const float4* hp = (const float4*)(h + (size_t)s * HID + c * 16);
  float* ap = agg + (size_t)d * HID + c * 16;
#pragma unroll
  for (int j = 0; j < 4; ++j) {
    float4 v = hp[j];
    atomicAdd(&ap[4 * j + 0], v.x * norm);
    atomicAdd(&ap[4 * j + 1], v.y * norm);
    atomicAdd(&ap[4 * j + 2], v.z * norm);
    atomicAdd(&ap[4 * j + 3], v.w * norm);
  }
}

__global__ void bias_relu_kernel(float* __restrict__ h, const float* __restrict__ b,
                                 long long n) {
  long long i = (long long)blockIdx.x * blockDim.x + threadIdx.x;
  if (i < n) {
    float v = h[i] + b[i & (HID - 1)];
    h[i] = v > 0.0f ? v : 0.0f;
  }
}

// ---------------- pair scoring: out[p] = h[i0].Wh[0:64] + h[i1].Wh[64:128] + bh ----------------

__global__ void pair_kernel(const float* __restrict__ h, const int* __restrict__ pairs,
                            const float* __restrict__ Wh, const float* __restrict__ bh,
                            float* __restrict__ out, int P) {
  int p = blockIdx.x * blockDim.x + threadIdx.x;
  if (p >= P) return;
  int i0 = pairs[2 * p];
  int i1 = pairs[2 * p + 1];
  const float4* r0 = (const float4*)(h + (size_t)i0 * HID);
  const float4* r1 = (const float4*)(h + (size_t)i1 * HID);
  const float4* w0 = (const float4*)Wh;
  const float4* w1 = (const float4*)(Wh + HID);
  float acc = bh[0];
#pragma unroll
  for (int j = 0; j < HID / 4; ++j) {
    float4 a = r0[j], wa = w0[j];
    float4 b = r1[j], wb = w1[j];
    acc += a.x * wa.x + a.y * wa.y + a.z * wa.z + a.w * wa.w;
    acc += b.x * wb.x + b.y * wb.y + b.z * wb.z + b.w * wb.w;
  }
  out[p] = acc;
}

// ---------------- launch ----------------

extern "C" void kernel_launch(void* const* d_in, const int* in_sizes, int n_in,
                              void* d_out, int out_size, void* d_ws, size_t ws_size,
                              hipStream_t stream) {
  const float* x     = (const float*)d_in[0];
  const int*   eidx  = (const int*)d_in[1];
  // d_in[2] = batch (unused by the reference output)
  const int*   pairs = (const int*)d_in[3];
  const float* W1    = (const float*)d_in[4];
  const float* b1    = (const float*)d_in[5];
  const float* W2    = (const float*)d_in[6];
  const float* b2    = (const float*)d_in[7];
  const float* Wh    = (const float*)d_in[8];
  const float* bh    = (const float*)d_in[9];
  float*       out   = (float*)d_out;

  const int       IN = 128;
  const int       N  = in_sizes[0] / IN;
  const long long E  = (long long)in_sizes[1] / 2;
  const int       P  = in_sizes[3] / 2;
  const int*      esrc = eidx;
  const int*      edst = eidx + E;

  const long long n64 = (long long)N * HID;

  // workspace: [isq: N f32][bufA: N*64 f32][bufB: N*64 f32][Wt1: 64x128 f16][Wt2: 64x64 f16]
  char* ws = (char*)d_ws;
  float*     isq  = (float*)ws;
  float*     bufA = (float*)(ws + (((size_t)N * 4 + 255) & ~(size_t)255));
  float*     bufB = bufA + n64;
  _Float16*  Wt1  = (_Float16*)(bufB + n64);
  _Float16*  Wt2  = Wt1 + (size_t)HID * IN;

  const int T = 256;
  int blkN    = (N + T - 1) / T;
  int blkE    = (int)((E + T - 1) / T);
  long long aggThreads = (E + (long long)N) * 4;
  int blkAgg  = (int)((aggThreads + T - 1) / T);
  int blkF    = (int)((n64 + T - 1) / T);
  int waves   = (N + 15) / 16;
  int blkGemm = (waves + (T / 32) - 1) / (T / 32);
  int blkP    = (P + T - 1) / T;
  int blkW1   = (HID * IN + T - 1) / T;
  int blkW2   = (HID * HID + T - 1) / T;

  // ---- weight prep (tiny, once per call) ----
  convert_wt_kernel<<<blkW1, T, 0, stream>>>(W1, Wt1, IN);
  convert_wt_kernel<<<blkW2, T, 0, stream>>>(W2, Wt2, HID);

  // ---- degree: deg = 1 (self-loop) + scatter, then rsqrt ----
  fill_kernel<<<blkN, T, 0, stream>>>(isq, 1.0f, N);
  deg_scatter_kernel<<<blkE, T, 0, stream>>>(edst, isq, E);
  rsqrt_kernel<<<blkN, T, 0, stream>>>(isq, N);

  // ---- layer 1: h1pre = x @ W1 (WMMA), aggregate, bias+relu ----
  gemm_wmma_kernel<<<blkGemm, T, 0, stream>>>(x, Wt1, bufA, N, IN);
  fill_kernel<<<blkF, T, 0, stream>>>(bufB, 0.0f, n64);
  agg_kernel<<<blkAgg, T, 0, stream>>>(bufA, esrc, edst, isq, bufB, E, N);
  bias_relu_kernel<<<blkF, T, 0, stream>>>(bufB, b1, n64);

  // ---- layer 2: h2pre = h1 @ W2 (WMMA), aggregate, bias+relu ----
  gemm_wmma_kernel<<<blkGemm, T, 0, stream>>>(bufB, Wt2, bufA, N, HID);
  fill_kernel<<<blkF, T, 0, stream>>>(bufB, 0.0f, n64);
  agg_kernel<<<blkAgg, T, 0, stream>>>(bufA, esrc, edst, isq, bufB, E, N);
  bias_relu_kernel<<<blkF, T, 0, stream>>>(bufB, b2, n64);

  // ---- pair scoring ----
  pair_kernel<<<blkP, T, 0, stream>>>(bufB, pairs, Wh, bh, out, P);
}
